// EncoderTransformerBlock2D_89558658056371
// MI455X (gfx1250) — matrix operand order repaired and checked by reading, not compiled, not run on hardware
//
#include <hip/hip_runtime.h>
#include <hip/hip_bf16.h>
#include <math.h>

typedef _Float16 half_t;
typedef __attribute__((ext_vector_type(16))) _Float16 v16h;
typedef __attribute__((ext_vector_type(8)))  _Float16 v8h;
typedef __attribute__((ext_vector_type(8)))  float    v8f;

#define TOK    262144   // B*H*W = 16*128*128
#define DIMC   192
#define HEADS  6
#define HD     32
#define NTOK   64       // tokens per window
#define NWIN   256
#define SHIFTV 4
#define SCALEQ 0.17677669529663687f  // 32^-0.5

// ---------------- WMMA fragment loaders (ISA 7.12.2 layouts) ----------------
// 16-bit A 16x32: lanes 0-15 row=base+lane, K = {k0..k0+7, k0+16..k0+23}
//                 lanes16-31 row=base+lane-16, K = {k0+8..k0+15, k0+24..k0+31}
__device__ inline v16h load_frag_a(const half_t* __restrict__ A, int lda,
                                   int row_base, int k0, int lane) {
  int row = row_base + (lane & 15);
  int kb  = k0 + ((lane >> 4) << 3);
  const half_t* p = A + (size_t)row * lda + kb;
  v8h lo = *(const v8h*)p;
  v8h hi = *(const v8h*)(p + 16);
  v16h r;
#pragma unroll
  for (int i = 0; i < 8; ++i) { r[i] = lo[i]; r[i + 8] = hi[i]; }
  return r;
}
// 16-bit B 32x16 (B given as Wt[N][K], K contiguous):
// lanes 0-15 col=base+lane, K = {k0..k0+7, k0+16..k0+23} ; lanes 16-31 col=base+lane-16, K = {k0+8..k0+15, k0+24..k0+31}
__device__ inline v16h load_frag_b(const half_t* __restrict__ Wt, int ldb,
                                   int col_base, int k0, int lane) {
  int col = col_base + (lane & 15);
  int kb  = k0 + ((lane >> 4) << 3);
  const half_t* p = Wt + (size_t)col * ldb + kb;
  v8h lo = *(const v8h*)p;
  v8h hi = *(const v8h*)(p + 16);
  v16h r;
#pragma unroll
  for (int i = 0; i < 8; ++i) { r[i] = lo[i]; r[i + 8] = hi[i]; }
  return r;
}

// ---------------- weight convert: f32 [K][N] -> f16 [N][K] ------------------
__global__ __launch_bounds__(256)
void convert_weights(const float* __restrict__ qw, const float* __restrict__ kvw,
                     const float* __restrict__ qb, const float* __restrict__ kvb,
                     const float* __restrict__ pw, const float* __restrict__ f1w,
                     const float* __restrict__ f2w,
                     half_t* __restrict__ qkvt, half_t* __restrict__ projt,
                     half_t* __restrict__ fc1t, half_t* __restrict__ fc2t,
                     float* __restrict__ qkvb) {
  int i = blockIdx.x * blockDim.x + threadIdx.x;
  if (i < 110592) {                       // qkv: 576 cols x 192
    int n = i / 192, k = i % 192;
    float s = (n < 192) ? qw[k * 192 + n] : kvw[k * 384 + (n - 192)];
    qkvt[n * 192 + k] = (half_t)s;
  } else if (i < 147456) {                // proj: 192 x 192
    int j = i - 110592; int n = j / 192, k = j % 192;
    projt[n * 192 + k] = (half_t)pw[k * 192 + n];
  } else if (i < 294912) {                // fc1: 768 cols x 192
    int j = i - 147456; int n = j / 192, k = j % 192;
    fc1t[n * 192 + k] = (half_t)f1w[k * 768 + n];
  } else if (i < 442368) {                // fc2: 192 cols x 768
    int j = i - 294912; int n = j / 768, k = j % 768;
    fc2t[n * 768 + k] = (half_t)f2w[k * 192 + n];
  } else if (i < 442944) {                // qkv bias concat
    int j = i - 442368;
    qkvb[j] = (j < 192) ? qb[j] : kvb[j - 192];
  }
}

// ---------------- LayerNorm (wave per token, optional window gather) --------
__global__ __launch_bounds__(256)
void ln_kernel(const float* __restrict__ xin, const float* __restrict__ g,
               const float* __restrict__ bta, half_t* __restrict__ out,
               int windowed) {
  int lane = threadIdx.x & 31;
  int wave = threadIdx.x >> 5;
  size_t m = (size_t)blockIdx.x * 8 + wave;
  const float* px;
  if (windowed) {
    int b_ = (int)(m >> 6), nt = (int)(m & 63);
    int bb = b_ >> 8, wi = b_ & 255, wh = wi >> 4, ww = wi & 15;
    int hh = ((wh << 3) + (nt >> 3) + SHIFTV) & 127;
    int wc = ((ww << 3) + (nt & 7) + SHIFTV) & 127;
    px = xin + (((size_t)bb * 128 + hh) * 128 + wc) * DIMC;
  } else {
    px = xin + m * DIMC;
  }
  float v[6], s = 0.f, s2 = 0.f;
#pragma unroll
  for (int j = 0; j < 6; ++j) {
    v[j] = px[lane + 32 * j]; s += v[j]; s2 += v[j] * v[j];
  }
#pragma unroll
  for (int off = 16; off; off >>= 1) {
    s  += __shfl_xor(s,  off, 32);
    s2 += __shfl_xor(s2, off, 32);
  }
  float mean = s * (1.f / 192.f);
  float var  = s2 * (1.f / 192.f) - mean * mean;
  float inv  = rsqrtf(var + 1e-5f);
#pragma unroll
  for (int j = 0; j < 6; ++j) {
    int c = lane + 32 * j;
    out[m * DIMC + c] = (half_t)((v[j] - mean) * inv * g[c] + bta[c]);
  }
}

// ---------------- generic WMMA GEMM with fused epilogues --------------------
// MODE 0: QKV scatter (+scale on q)   MODE 1: proj + un-window/un-shift + residual
// MODE 2: fc1 + exact GELU            MODE 3: fc2 + residual -> final output
template <int MODE>
__global__ __launch_bounds__(256)
void gemm_wmma(const half_t* __restrict__ A, const half_t* __restrict__ Wt,
               const float* __restrict__ bias, int K, int N,
               half_t* __restrict__ oq, half_t* __restrict__ ok,
               half_t* __restrict__ ov, const float* __restrict__ fin,
               float* __restrict__ fout) {
  const int tid  = threadIdx.x;
  const int lane = tid & 31;
  const int wave = tid >> 5;
  const int tile_m = blockIdx.y * 32 + (wave >> 2) * 16;
  const int tile_n = blockIdx.x * 64 + (wave & 3) * 16;

  v8f acc;
  {
    float bv = bias[tile_n + (lane & 15)];
#pragma unroll
    for (int i = 0; i < 8; ++i) acc[i] = bv;
  }
  for (int k0 = 0; k0 < K; k0 += 32) {
    v16h a = load_frag_a(A,  K, tile_m, k0, lane);
    v16h b = load_frag_b(Wt, K, tile_n, k0, lane);
    acc = __builtin_amdgcn_wmma_f32_16x16x32_f16(false, a, false, b,
                                                 (short)0, acc, false, false);
  }
  const int col  = tile_n + (lane & 15);
  const int rowb = tile_m + ((lane >> 4) << 3);

  if (MODE == 0) {  // q/k/v scatter to (win, head, tok, d); fold softmax scale into q
#pragma unroll
    for (int r = 0; r < 8; ++r) {
      int m = rowb + r;
      size_t b_ = (size_t)(m >> 6); int nt = m & 63;
      float val = acc[r];
      if (col < DIMC) {
        int h = col >> 5, d = col & 31;
        oq[((b_ * HEADS + h) * NTOK + nt) * HD + d] = (half_t)(val * SCALEQ);
      } else if (col < 2 * DIMC) {
        int c2 = col - DIMC; int h = c2 >> 5, d = c2 & 31;
        ok[((b_ * HEADS + h) * NTOK + nt) * HD + d] = (half_t)val;
      } else {
        int c2 = col - 2 * DIMC; int h = c2 >> 5, d = c2 & 31;
        ov[((b_ * HEADS + h) * NTOK + nt) * HD + d] = (half_t)val;
      }
    }
  } else if (MODE == 1) {  // un-window + un-shift + residual -> x1 (f32 BHWC)
#pragma unroll
    for (int r = 0; r < 8; ++r) {
      int m = rowb + r;
      int b_ = m >> 6, nt = m & 63;
      int bb = b_ >> 8, wi = b_ & 255, wh = wi >> 4, ww = wi & 15;
      int hh = ((wh << 3) + (nt >> 3) + SHIFTV) & 127;
      int wc = ((ww << 3) + (nt & 7) + SHIFTV) & 127;
      size_t idx = (((size_t)bb * 128 + hh) * 128 + wc) * DIMC + col;
      fout[idx] = fin[idx] + acc[r];
    }
  } else if (MODE == 2) {  // exact GELU
#pragma unroll
    for (int r = 0; r < 8; ++r) {
      int m = rowb + r;
      float xg = acc[r];
      float ge = 0.5f * xg * (1.f + erff(xg * 0.70710678118654752f));
      oq[(size_t)m * N + col] = (half_t)ge;
    }
  } else {                 // MODE 3: residual -> final f32 output
#pragma unroll
    for (int r = 0; r < 8; ++r) {
      size_t idx = (size_t)(rowb + r) * DIMC + col;
      fout[idx] = fin[idx] + acc[r];
    }
  }
}

// ---------------- attention: one block per (window, head) ------------------
__global__ __launch_bounds__(128)
void attn_kernel(const half_t* __restrict__ q, const half_t* __restrict__ k,
                 const half_t* __restrict__ v, const float* __restrict__ rpb,
                 const float* __restrict__ mask, half_t* __restrict__ ao) {
  __shared__ __align__(16) half_t sQ[NTOK * HD];
  __shared__ __align__(16) half_t sK[NTOK * HD];
  __shared__ __align__(16) half_t sVt[HD * NTOK];   // V transposed: [d][tok]
  __shared__ __align__(16) float  sS[NTOK * NTOK];
  __shared__ __align__(16) half_t sP[NTOK * NTOK];

  const int b_  = blockIdx.x;
  const int h   = blockIdx.y;
  const int tid = threadIdx.x;
  const size_t base = ((size_t)b_ * HEADS + h) * (NTOK * HD);

  *(v8h*)(sQ + tid * 16)     = *(const v8h*)(q + base + tid * 16);
  *(v8h*)(sQ + tid * 16 + 8) = *(const v8h*)(q + base + tid * 16 + 8);
  *(v8h*)(sK + tid * 16)     = *(const v8h*)(k + base + tid * 16);
  *(v8h*)(sK + tid * 16 + 8) = *(const v8h*)(k + base + tid * 16 + 8);
#pragma unroll
  for (int i = 0; i < 16; ++i) {
    int e = tid * 16 + i;
    sVt[(e & 31) * NTOK + (e >> 5)] = v[base + e];
  }
  __syncthreads();

  const int lane = tid & 31;
  const int wave = tid >> 5;
  const size_t mbase = (size_t)(b_ & 255) * (NTOK * NTOK);

  // S = q*K^T (+rel-pos bias, +swin mask); head_dim=32 -> single WMMA per tile
#pragma unroll
  for (int i = 0; i < 4; ++i) {
    int t  = wave * 4 + i;
    int tm = (t >> 2) << 4;
    int tn = (t & 3)  << 4;
    v16h a = load_frag_a(sQ, HD, tm, 0, lane);
    v16h b = load_frag_b(sK, HD, tn, 0, lane);
    v8f acc = {};
    acc = __builtin_amdgcn_wmma_f32_16x16x32_f16(false, a, false, b,
                                                 (short)0, acc, false, false);
    int col  = tn + (lane & 15);
    int rowb = tm + ((lane >> 4) << 3);
    int rn = col >> 3, cn = col & 7;
#pragma unroll
    for (int r = 0; r < 8; ++r) {
      int m = rowb + r;
      int ridx = ((m >> 3) - rn + 7) * 15 + ((m & 7) - cn + 7);
      sS[m * NTOK + col] =
          acc[r] + rpb[ridx * HEADS + h] + mask[mbase + m * NTOK + col];
    }
  }
  __syncthreads();

  if (tid < NTOK) {  // row softmax
    float mx = -1e30f;
#pragma unroll 8
    for (int j = 0; j < NTOK; ++j) mx = fmaxf(mx, sS[tid * NTOK + j]);
    float sum = 0.f;
#pragma unroll 8
    for (int j = 0; j < NTOK; ++j) {
      float e = __expf(sS[tid * NTOK + j] - mx);
      sS[tid * NTOK + j] = e;
      sum += e;
    }
    float inv = 1.f / sum;
#pragma unroll 8
    for (int j = 0; j < NTOK; ++j)
      sP[tid * NTOK + j] = (half_t)(sS[tid * NTOK + j] * inv);
  }
  __syncthreads();

  // out = P @ V  (64x32, K=64)
#pragma unroll
  for (int i = 0; i < 2; ++i) {
    int tm = wave << 4;
    int tn = i << 4;
    v8f acc = {};
#pragma unroll
    for (int k0 = 0; k0 < NTOK; k0 += 32) {
      v16h a = load_frag_a(sP,  NTOK, tm, k0, lane);
      v16h b = load_frag_b(sVt, NTOK, tn, k0, lane);
      acc = __builtin_amdgcn_wmma_f32_16x16x32_f16(false, a, false, b,
                                                   (short)0, acc, false, false);
    }
    int d    = tn + (lane & 15);
    int rowb = tm + ((lane >> 4) << 3);
#pragma unroll
    for (int r = 0; r < 8; ++r) {
      int m = rowb + r;
      ao[((size_t)b_ * NTOK + m) * DIMC + h * HD + d] = (half_t)acc[r];
    }
  }
}

// ---------------------------------------------------------------------------
extern "C" void kernel_launch(void* const* d_in, const int* in_sizes, int n_in,
                              void* d_out, int out_size, void* d_ws,
                              size_t ws_size, hipStream_t stream) {
  const float* x    = (const float*)d_in[0];
  const float* mask = (const float*)d_in[1];
  const float* n1g  = (const float*)d_in[2];
  const float* n1b  = (const float*)d_in[3];
  const float* qw   = (const float*)d_in[4];
  const float* qb   = (const float*)d_in[5];
  const float* kvw  = (const float*)d_in[6];
  const float* kvb  = (const float*)d_in[7];
  const float* rpb  = (const float*)d_in[8];
  const float* pw   = (const float*)d_in[9];
  const float* pb   = (const float*)d_in[10];
  const float* n2g  = (const float*)d_in[11];
  const float* n2b  = (const float*)d_in[12];
  const float* f1w  = (const float*)d_in[13];
  const float* f1b  = (const float*)d_in[14];
  const float* f2w  = (const float*)d_in[15];
  const float* f2b  = (const float*)d_in[16];
  float* out = (float*)d_out;
  char*  ws  = (char*)d_ws;

  const size_t nE = (size_t)TOK * DIMC;  // 50,331,648 elements

  half_t* qkvt  = (half_t*)ws;           // [576][192] f16
  half_t* projt = qkvt + 110592;         // [192][192]
  half_t* fc1t  = projt + 36864;         // [768][192]
  half_t* fc2t  = fc1t + 147456;         // [192][768]
  float*  qkvb  = (float*)(ws + 884736); // 576 f32
  half_t* xw    = (half_t*)(ws + (1u << 20));         // LN1 windowed, f16
  half_t* qbuf  = (half_t*)((char*)xw + nE * 2);
  half_t* kbuf  = qbuf + nE;
  half_t* vbuf  = kbuf + nE;
  half_t* aobuf = vbuf + nE;                           // attn out, f16
  float*  x1    = (float*)((char*)aobuf + nE * 2);     // residual1, f32
  half_t* h16   = (half_t*)((char*)x1 + nE * 4);       // LN2 out, f16
  half_t* h2    = h16 + nE;                            // GELU out, f16 (TOK*768)

  convert_weights<<<(442944 + 255) / 256, 256, 0, stream>>>(
      qw, kvw, qb, kvb, pw, f1w, f2w, qkvt, projt, fc1t, fc2t, qkvb);

  ln_kernel<<<TOK / 8, 256, 0, stream>>>(x, n1g, n1b, xw, 1);

  gemm_wmma<0><<<dim3(576 / 64, TOK / 32), 256, 0, stream>>>(
      xw, qkvt, qkvb, 192, 576, qbuf, kbuf, vbuf, nullptr, nullptr);

  attn_kernel<<<dim3(4096, HEADS), 128, 0, stream>>>(qbuf, kbuf, vbuf, rpb,
                                                     mask, aobuf);

  gemm_wmma<1><<<dim3(192 / 64, TOK / 32), 256, 0, stream>>>(
      aobuf, projt, pb, 192, 192, nullptr, nullptr, nullptr, x, x1);

  ln_kernel<<<TOK / 8, 256, 0, stream>>>(x1, n2g, n2b, h16, 0);

  gemm_wmma<2><<<dim3(768 / 64, TOK / 32), 256, 0, stream>>>(
      h16, fc1t, f1b, 192, 768, h2, nullptr, nullptr, nullptr, nullptr);

  gemm_wmma<3><<<dim3(192 / 64, TOK / 32), 256, 0, stream>>>(
      h2, fc2t, f2b, 768, 192, nullptr, nullptr, nullptr, x1, out);
}
